// WRATBlock_81252191306592
// MI455X (gfx1250) — matrix operand order repaired and checked by reading, not compiled, not run on hardware
//
#include <hip/hip_runtime.h>

// ---------------- CDNA5 (gfx1250, wave32) WMMA transformer block ----------------
// All GEMMs + attention run on v_wmma_f32_16x16x32_bf16. bf16 storage, f32 accum.
// GEMM: 32x64 tile/wave, double-buffered K loop (loads k+1 issued before wmmas of k).

typedef __attribute__((ext_vector_type(16))) __bf16   v16bf;
typedef __attribute__((ext_vector_type(8)))  float    v8f;
typedef __attribute__((ext_vector_type(4)))  unsigned u32x4;
typedef __attribute__((ext_vector_type(4)))  float    f32x4;

#define Dm 512
#define Ls 2048
#define Bn 2
#define Hn 8
#define DH 64
#define Mt (Bn*Ls)      // 4096 rows
#define Fh (4*Dm)       // 2048
#define TAU 0.1f
#define VTS 40          // padded LDS stride (halves) for V^T tile

union BFPack {
  v16bf v;
  u32x4 q[2];
  unsigned u[8];
  unsigned short s[16];
};

static __device__ __forceinline__ v8f v8zero() {
  v8f z = {0.f,0.f,0.f,0.f,0.f,0.f,0.f,0.f};
  return z;
}
static __device__ __forceinline__ v8f wmma32_bf16(v16bf a, v16bf b, v8f c) {
  return __builtin_amdgcn_wmma_f32_16x16x32_bf16(false, a, false, b, (short)0, c,
                                                 false, false);
}
static __device__ __forceinline__ unsigned short f2bf(float x) {
  union { float f; unsigned u; } t; t.f = x;
  unsigned r = t.u + 0x7fffu + ((t.u >> 16) & 1u);   // round-to-nearest-even
  return (unsigned short)(r >> 16);
}

// ---------------- f32 -> bf16 weight conversion ----------------
__global__ void __launch_bounds__(256) f2bf_k(const float* __restrict__ s,
                                              unsigned short* __restrict__ d, int n) {
  int i = blockIdx.x * 256 + threadIdx.x;
  if (i < n) d[i] = f2bf(s[i]);
}

// ---------------- LDS-tiled transpose: src[z][R][C] -> dst[z][C][R] ----------------
__global__ void __launch_bounds__(256) transpose_rc(const float* __restrict__ src,
                                                    float* __restrict__ dst,
                                                    unsigned short* __restrict__ dstB,
                                                    int R, int C) {
  __shared__ float tile[32][33];
  size_t zoff = (size_t)blockIdx.z * R * C;
  int c0 = blockIdx.x * 32, r0 = blockIdx.y * 32;
  int tx = threadIdx.x, ty = threadIdx.y;
#pragma unroll
  for (int i = 0; i < 4; ++i)
    tile[ty + i*8][tx] = src[zoff + (size_t)(r0 + ty + i*8) * C + c0 + tx];
  __syncthreads();
#pragma unroll
  for (int i = 0; i < 4; ++i) {
    float v = tile[tx][ty + i*8];
    size_t o = zoff + (size_t)(c0 + ty + i*8) * R + r0 + tx;
    dst[o] = v;
    if (dstB) dstB[o] = f2bf(v);
  }
}

// ---------------- energy mask: mean_d |x[row][d]| > tau (wave per row) ----------------
__global__ void __launch_bounds__(256) energy_mask_k(const float* __restrict__ X,
                                                     float* __restrict__ mask) {
  int row  = blockIdx.x * 8 + (threadIdx.x >> 5);
  int lane = threadIdx.x & 31;
  size_t base = (size_t)row * Dm;
  float s = 0.f;
#pragma unroll
  for (int i = 0; i < 4; ++i) {
    f32x4 a = *(const f32x4*)(X + base + i*128 + lane*4);
    s += fabsf(a[0]) + fabsf(a[1]) + fabsf(a[2]) + fabsf(a[3]);
  }
#pragma unroll
  for (int off = 16; off >= 1; off >>= 1) s += __shfl_xor(s, off);
  if (lane == 0) mask[row] = (s * (1.f/(float)Dm) > TAU) ? 1.f : 0.f;
}

// ---------------- WMMA GEMM: Y[M,N] = A[M,K] @ W[N,K]^T + bias, fused epilogue -----
// wave -> 32x64 tile (2 M x 4 N subtiles); ping-pong double buffering on K.
__global__ void __launch_bounds__(256) gemm_bf16(
    const unsigned short* __restrict__ A, const unsigned short* __restrict__ W,
    const float* __restrict__ bias, int M, int N, int K,
    float* __restrict__ outF, unsigned short* __restrict__ outB,
    const float* __restrict__ resid, int act) {
  int wid  = blockIdx.x * 8 + (threadIdx.x >> 5);
  int lane = threadIdx.x & 31;
  int ln = lane & 15, h2 = lane >> 4;
  int tilesN = N >> 6;
  int tm = wid / tilesN, tn = wid % tilesN;
  const unsigned short* arow  = A + (size_t)(tm*32 + ln) * K;   // +mI*16*K per subtile
  const unsigned short* wbase = W + (size_t)(tn*64 + ln) * K;   // +j*16*K per subtile

  v8f acc[2][4];
#pragma unroll
  for (int mI = 0; mI < 2; ++mI)
#pragma unroll
    for (int j = 0; j < 4; ++j) acc[mI][j] = v8zero();

  BFPack abuf[2][2], bbuf[2][4];
  auto load_step = [&](BFPack (&a)[2], BFPack (&b)[4], int k0) {
#pragma unroll
    for (int mI = 0; mI < 2; ++mI) {            // A: K interleave-16 (contiguous 32B)
      const unsigned short* ar = arow + (size_t)mI*16*K + k0;
      a[mI].q[0] = *(const u32x4*)(ar + h2*16);
      a[mI].q[1] = *(const u32x4*)(ar + h2*16 + 8);
    }
#pragma unroll
    for (int j = 0; j < 4; ++j) {               // B: K interleave-16 (contiguous 32B)
      const unsigned short* wr = wbase + (size_t)j*16*K + k0 + h2*16;
      b[j].q[0] = *(const u32x4*)(wr);
      b[j].q[1] = *(const u32x4*)(wr + 8);
    }
  };
  auto math_step = [&](BFPack (&a)[2], BFPack (&b)[4]) {
#pragma unroll
    for (int j = 0; j < 4; ++j)
#pragma unroll
      for (int mI = 0; mI < 2; ++mI)
        acc[mI][j] = wmma32_bf16(a[mI].v, b[j].v, acc[mI][j]);
  };

  load_step(abuf[0], bbuf[0], 0);
  for (int k0 = 0; k0 < K; k0 += 64) {
    __builtin_prefetch(arow + k0 + 128, 0, 1);            // global_prefetch_b8
    __builtin_prefetch(wbase + k0 + 128, 0, 1);
    load_step(abuf[1], bbuf[1], k0 + 32);
    math_step(abuf[0], bbuf[0]);
    if (k0 + 64 < K) load_step(abuf[0], bbuf[0], k0 + 64);
    math_step(abuf[1], bbuf[1]);
  }

#pragma unroll
  for (int mI = 0; mI < 2; ++mI)
#pragma unroll
    for (int j = 0; j < 4; ++j) {
      int c = tn*64 + j*16 + ln;
      float bb = bias ? bias[c] : 0.f;
#pragma unroll
      for (int v = 0; v < 8; ++v) {
        int r = tm*32 + mI*16 + v + h2*8;       // C/D layout: M = vgpr + 8*(lane/16)
        float y = acc[mI][j][v] + bb;
        if (act) y = 0.5f * y * (1.f + erff(y * 0.70710678118654752f)); // exact GELU
        if (resid) y += resid[(size_t)r * N + c];
        if (outF) outF[(size_t)r * N + c] = y;
        if (outB) outB[(size_t)r * N + c] = f2bf(y);
      }
    }
}

// ---------------- flash attention (wave: 16 q rows; block: 4 waves = 64 q rows) ----
// S^T = K @ Q^T  (q in lanes -> softmax reduce = in-reg + one xor16 shuffle)
// O^T = V^T @ P  (V^T staged transposed in LDS; P^T regs -> B operand via xor16)
__global__ void __launch_bounds__(128) attn_wmma(
    const unsigned short* __restrict__ Q, const unsigned short* __restrict__ Kmat,
    const unsigned short* __restrict__ V, const float* __restrict__ mask,
    unsigned short* __restrict__ ctx) {
  __shared__ unsigned short VT[64 * VTS];
  int bid = blockIdx.x;
  int qb = bid & 31, h = (bid >> 5) & 7, b = bid >> 8;
  int wid = threadIdx.x >> 5, lane = threadIdx.x & 31;
  int ln = lane & 15, h2 = lane >> 4;
  int q0 = qb * 64 + wid * 16;

  size_t rowQ = (size_t)(b*Ls + q0 + ln) * Dm + h*DH;
  BFPack bq[2];                                  // Q^T B-operands for both d-chunks
#pragma unroll
  for (int c = 0; c < 2; ++c) {
    bq[c].q[0] = *(const u32x4*)(Q + rowQ + c*32 + h2*16);
    bq[c].q[1] = *(const u32x4*)(Q + rowQ + c*32 + h2*16 + 8);
  }

  v8f o[4];
#pragma unroll
  for (int t = 0; t < 4; ++t) o[t] = v8zero();
  float m = -INFINITY, l = 0.f;
  const float* maskb = mask ? (mask + (size_t)b * Ls) : (const float*)0;

  for (int kb = 0; kb < Ls; kb += 32) {
    // prefetch next V chunk while we work on this one
    __builtin_prefetch(V + (size_t)(b*Ls + kb + 32 + (threadIdx.x >> 2)) * Dm + h*DH,
                       0, 1);
    __syncthreads();
    {   // cooperative stage: V[kb..kb+32)[h*64..+64) -> VT[dh][k] (transposed)
      int krow = threadIdx.x >> 2, seg = (threadIdx.x & 3) * 16;
      const unsigned short* src = V + (size_t)(b*Ls + kb + krow) * Dm + h*DH + seg;
      BFPack stg;
      stg.q[0] = *(const u32x4*)(src);
      stg.q[1] = *(const u32x4*)(src + 8);
#pragma unroll
      for (int j = 0; j < 16; ++j) VT[(seg + j) * VTS + krow] = stg.s[j];
    }
    __syncthreads();

    v8f st[2];
#pragma unroll
    for (int t = 0; t < 2; ++t) {                // two 16-key S^T tiles
      v8f s = v8zero();
      const unsigned short* krp = Kmat + (size_t)(b*Ls + kb + t*16 + ln) * Dm + h*DH;
#pragma unroll
      for (int c = 0; c < 2; ++c) {
        BFPack a;
        a.q[0] = *(const u32x4*)(krp + c*32 + h2*16);
        a.q[1] = *(const u32x4*)(krp + c*32 + h2*16 + 8);
        s = wmma32_bf16(a.v, bq[c].v, s);
      }
#pragma unroll
      for (int v = 0; v < 8; ++v) s[v] *= 0.125f;   // 1/sqrt(Dh)
      if (maskb) {
        f32x4 m0 = *(const f32x4*)(maskb + kb + t*16 + h2*8);
        f32x4 m1 = *(const f32x4*)(maskb + kb + t*16 + h2*8 + 4);
#pragma unroll
        for (int v = 0; v < 4; ++v) { if (m0[v] == 0.f) s[v]   = -INFINITY; }
#pragma unroll
        for (int v = 0; v < 4; ++v) { if (m1[v] == 0.f) s[v+4] = -INFINITY; }
      }
      st[t] = s;
    }

    // ---- online softmax (per q column = per lane, halves combined via xor16) ----
    float cmax = -INFINITY;
#pragma unroll
    for (int v = 0; v < 8; ++v) cmax = fmaxf(cmax, fmaxf(st[0][v], st[1][v]));
    cmax = fmaxf(cmax, __shfl_xor(cmax, 16));
    float nm = fmaxf(m, cmax);
    float alpha = (m == -INFINITY) ? 0.f : __expf(m - nm);
    float p0[8], p1[8], ps = 0.f;
#pragma unroll
    for (int v = 0; v < 8; ++v) {
      p0[v] = (nm == -INFINITY) ? 0.f : __expf(st[0][v] - nm);
      p1[v] = (nm == -INFINITY) ? 0.f : __expf(st[1][v] - nm);
      ps += p0[v] + p1[v];
    }
    ps += __shfl_xor(ps, 16);
    l = l * alpha + ps;
    m = nm;
#pragma unroll
    for (int t = 0; t < 4; ++t)
#pragma unroll
      for (int v = 0; v < 8; ++v) o[t][v] *= alpha;

    // ---- pack P^T accum -> B operand (K interleave-16) via one xor16 exchange ----
    unsigned pa[4], pb[4], ra[4], rb[4];
#pragma unroll
    for (int i = 0; i < 4; ++i) {
      pa[i] = (unsigned)f2bf(p0[2*i]) | ((unsigned)f2bf(p0[2*i+1]) << 16);
      pb[i] = (unsigned)f2bf(p1[2*i]) | ((unsigned)f2bf(p1[2*i+1]) << 16);
    }
#pragma unroll
    for (int i = 0; i < 4; ++i) {
      ra[i] = (unsigned)__shfl_xor((int)pa[i], 16);
      rb[i] = (unsigned)__shfl_xor((int)pb[i], 16);
    }
    BFPack pB;
#pragma unroll
    for (int i = 0; i < 4; ++i) {
      pB.u[i]     = h2 ? rb[i] : pa[i];   // k rows h2*16 .. +7
      pB.u[i + 4] = h2 ? pb[i] : ra[i];   // k rows h2*16+8 .. +15
    }

    // ---- O^T += V^T @ P, four 16-wide dh tiles, A from LDS (ds_load_b128) ----
#pragma unroll
    for (int t = 0; t < 4; ++t) {
      const unsigned short* vp = VT + (t*16 + ln) * VTS;
      BFPack a;
      a.q[0] = *(const u32x4*)(vp + h2*16);
      a.q[1] = *(const u32x4*)(vp + h2*16 + 8);
      o[t] = wmma32_bf16(a.v, pB.v, o[t]);
    }
  }

  float rl = (l > 0.f) ? (1.f / l) : 0.f;        // all-masked rows -> 0 (nan_to_num)
  size_t crow = (size_t)(b*Ls + q0 + ln) * Dm + h*DH;
#pragma unroll
  for (int t = 0; t < 4; ++t)
#pragma unroll
    for (int i = 0; i < 4; ++i) {
      unsigned pk = (unsigned)f2bf(o[t][2*i] * rl) |
                    ((unsigned)f2bf(o[t][2*i+1] * rl) << 16);
      *(unsigned*)(ctx + crow + t*16 + h2*8 + 2*i) = pk;
    }
}

// ---------------- fused residual + LayerNorm (wave per row) ----------------
__global__ void __launch_bounds__(256) ln_fuse(
    const float* __restrict__ A, const float* __restrict__ Bv,
    const float* __restrict__ Cv, const float* __restrict__ g,
    const float* __restrict__ be, float* __restrict__ outF,
    unsigned short* __restrict__ outB) {
  int row  = blockIdx.x * 8 + (threadIdx.x >> 5);
  int lane = threadIdx.x & 31;
  size_t base = (size_t)row * Dm;
  float x[16];
#pragma unroll
  for (int i = 0; i < 4; ++i) {
    int col = i*128 + lane*4;
    f32x4 a = *(const f32x4*)(A  + base + col);
    f32x4 b = *(const f32x4*)(Bv + base + col);
#pragma unroll
    for (int j = 0; j < 4; ++j) x[i*4+j] = a[j] + b[j];
    if (Cv) {
      f32x4 c = *(const f32x4*)(Cv + base + col);
#pragma unroll
      for (int j = 0; j < 4; ++j) x[i*4+j] += c[j];
    }
  }
  float s = 0.f, ss = 0.f;
#pragma unroll
  for (int i = 0; i < 16; ++i) { s += x[i]; ss += x[i]*x[i]; }
#pragma unroll
  for (int off = 16; off >= 1; off >>= 1) {
    s += __shfl_xor(s, off); ss += __shfl_xor(ss, off);
  }
  float mean = s * (1.f/(float)Dm);
  float var  = ss * (1.f/(float)Dm) - mean*mean;
  float rstd = rsqrtf(var + 1e-5f);
#pragma unroll
  for (int i = 0; i < 4; ++i)
#pragma unroll
    for (int j = 0; j < 4; ++j) {
      int col = i*128 + lane*4 + j;
      float y = (x[i*4+j] - mean) * rstd * g[col] + be[col];
      outF[base + col] = y;
      outB[base + col] = f2bf(y);
    }
}

// ================================ host ================================
extern "C" void kernel_launch(void* const* d_in, const int* in_sizes, int n_in,
                              void* d_out, int out_size, void* d_ws, size_t ws_size,
                              hipStream_t stream) {
  (void)in_sizes; (void)n_in; (void)out_size; (void)ws_size;
  typedef unsigned short ush;
  const float* LL = (const float*)d_in[0];
  const float* LH = (const float*)d_in[1];
  // params flattened: per attn {Wq,bq,Wk,bk,Wv,bv,Wo,bo}; mlp {W1,b1,W2,b2}; ln g/b
  const int aLL = 2, aLH = 10, aCr = 18, mLL = 26, mLH = 30;
  const float* ln1g = (const float*)d_in[34];
  const float* ln1b = (const float*)d_in[35];
  const float* ln2g = (const float*)d_in[36];
  const float* ln2b = (const float*)d_in[37];

  char* wsp = (char*)d_ws;
  auto carve = [&](size_t bytes) -> char* {
    char* p = wsp; wsp += (bytes + 255) & ~(size_t)255; return p;
  };
  const size_t MD = (size_t)Mt * Dm, MF = (size_t)Mt * Fh;
  float* xLLf = (float*)carve(MD*4);   float* xLHf = (float*)carve(MD*4);
  ush*   xLLb = (ush*)carve(MD*2);     ush*   xLHb = (ush*)carve(MD*2);
  float* maskLH = (float*)carve(Mt*4);
  ush* wA[3][4]; // bf16 attn weights [attn][q,k,v,o]
  for (int a = 0; a < 3; ++a)
    for (int i = 0; i < 4; ++i) wA[a][i] = (ush*)carve((size_t)Dm*Dm*2);
  ush* w1LL = (ush*)carve((size_t)Fh*Dm*2); ush* w2LL = (ush*)carve((size_t)Dm*Fh*2);
  ush* w1LH = (ush*)carve((size_t)Fh*Dm*2); ush* w2LH = (ush*)carve((size_t)Dm*Fh*2);
  ush* Qb  = (ush*)carve(MD*2); ush* Kbf = (ush*)carve(MD*2);
  ush* Vb  = (ush*)carve(MD*2); ush* ctx = (ush*)carve(MD*2);
  float* LLoF = (float*)carve(MD*4); ush* LLoB = (ush*)carve(MD*2);
  float* LHoF = (float*)carve(MD*4); ush* LHoB = (ush*)carve(MD*2);
  float* crF  = (float*)carve(MD*4);
  float* fLLf = (float*)carve(MD*4); ush* fLLb = (ush*)carve(MD*2);
  float* fLHf = (float*)carve(MD*4); ush* fLHb = (ush*)carve(MD*2);
  ush* hid = (ush*)carve(MF*2);
  float* finLL = xLLf;  // reuse: input seqs dead after LayerNorm
  float* finLH = xLHf;

  auto cv = [&](int idx, ush* dst, size_t n) {
    f2bf_k<<<(unsigned)((n + 255) / 256), 256, 0, stream>>>(
        (const float*)d_in[idx], dst, (int)n);
  };
  auto gemm = [&](const ush* A, const ush* W, const float* bias, int M, int N, int K,
                  float* oF, ush* oB, const float* resid, int act) {
    unsigned blocks = (unsigned)((M/32) * (N/64) / 8);
    gemm_bf16<<<blocks, 256, 0, stream>>>(A, W, bias, M, N, K, oF, oB, resid, act);
  };
  auto attn = [&](const ush* q, const ush* k, const ush* v, const float* msk, ush* o) {
    attn_wmma<<<Bn*Hn*(Ls/64), 128, 0, stream>>>(q, k, v, msk, o);
  };
  auto fp = [&](int idx) { return (const float*)d_in[idx]; };

  dim3 tb(32, 8);
  // 1) [B,D,L] -> [B,L,D] (f32 + bf16)
  transpose_rc<<<dim3(Ls/32, Dm/32, Bn), tb, 0, stream>>>(LL, xLLf, xLLb, Dm, Ls);
  transpose_rc<<<dim3(Ls/32, Dm/32, Bn), tb, 0, stream>>>(LH, xLHf, xLHb, Dm, Ls);
  // 2) energy mask for LH branch
  energy_mask_k<<<Mt/8, 256, 0, stream>>>(xLHf, maskLH);
  // 3) weights -> bf16
  const int ab[3] = {aLL, aLH, aCr};
  for (int a = 0; a < 3; ++a)
    for (int i = 0; i < 4; ++i) cv(ab[a] + 2*i, wA[a][i], (size_t)Dm*Dm);
  cv(mLL+0, w1LL, (size_t)Fh*Dm); cv(mLL+2, w2LL, (size_t)Dm*Fh);
  cv(mLH+0, w1LH, (size_t)Fh*Dm); cv(mLH+2, w2LH, (size_t)Dm*Fh);
  // 4) self-attention LL
  gemm(xLLb, wA[0][0], fp(aLL+1), Mt, Dm, Dm, nullptr, Qb, nullptr, 0);
  gemm(xLLb, wA[0][1], fp(aLL+3), Mt, Dm, Dm, nullptr, Kbf, nullptr, 0);
  gemm(xLLb, wA[0][2], fp(aLL+5), Mt, Dm, Dm, nullptr, Vb, nullptr, 0);
  attn(Qb, Kbf, Vb, nullptr, ctx);
  gemm(ctx, wA[0][3], fp(aLL+7), Mt, Dm, Dm, LLoF, LLoB, nullptr, 0);
  // 5) masked self-attention LH
  gemm(xLHb, wA[1][0], fp(aLH+1), Mt, Dm, Dm, nullptr, Qb, nullptr, 0);
  gemm(xLHb, wA[1][1], fp(aLH+3), Mt, Dm, Dm, nullptr, Kbf, nullptr, 0);
  gemm(xLHb, wA[1][2], fp(aLH+5), Mt, Dm, Dm, nullptr, Vb, nullptr, 0);
  attn(Qb, Kbf, Vb, maskLH, ctx);
  gemm(ctx, wA[1][3], fp(aLH+7), Mt, Dm, Dm, LHoF, LHoB, nullptr, 0);
  // 6) cross attention: Q from LL_out, K/V from LH_out
  gemm(LLoB, wA[2][0], fp(aCr+1), Mt, Dm, Dm, nullptr, Qb, nullptr, 0);
  gemm(LHoB, wA[2][1], fp(aCr+3), Mt, Dm, Dm, nullptr, Kbf, nullptr, 0);
  gemm(LHoB, wA[2][2], fp(aCr+5), Mt, Dm, Dm, nullptr, Vb, nullptr, 0);
  attn(Qb, Kbf, Vb, nullptr, ctx);
  gemm(ctx, wA[2][3], fp(aCr+7), Mt, Dm, Dm, crF, nullptr, nullptr, 0);
  // 7) residual + LayerNorm
  ln_fuse<<<Mt/8, 256, 0, stream>>>(xLLf, LLoF, crF, ln1g, ln1b, fLLf, fLLb);
  ln_fuse<<<Mt/8, 256, 0, stream>>>(xLHf, LHoF, nullptr, ln2g, ln2b, fLHf, fLHb);
  // 8) MLPs (GELU in first GEMM, residual in second)
  gemm(fLLb, w1LL, fp(mLL+1), Mt, Fh, Dm, nullptr, hid, nullptr, 1);
  gemm(hid,  w2LL, fp(mLL+3), Mt, Dm, Fh, finLL, nullptr, fLLf, 0);
  gemm(fLHb, w1LH, fp(mLH+1), Mt, Fh, Dm, nullptr, hid, nullptr, 1);
  gemm(hid,  w2LH, fp(mLH+3), Mt, Dm, Fh, finLH, nullptr, fLHf, 0);
  // 9) [B,L,D] -> [B,D,L] into d_out (LL then LH)
  float* outF = (float*)d_out;
  transpose_rc<<<dim3(Dm/32, Ls/32, Bn), tb, 0, stream>>>(finLL, outF, nullptr, Ls, Dm);
  transpose_rc<<<dim3(Dm/32, Ls/32, Bn), tb, 0, stream>>>(
      finLH, outF + (size_t)Bn*Dm*Ls, nullptr, Ls, Dm);
}
